// MultiHeadAttention_46712064311488
// MI455X (gfx1250) — matrix operand order repaired and checked by reading, not compiled, not run on hardware
//
#include <hip/hip_runtime.h>
#include <hip/hip_bf16.h>

// ---------------------------------------------------------------------------
// MHA forward for MI455X (gfx1250, wave32) built on v_wmma_f32_16x16x32_bf16.
// Pipeline: f32->bf16 conversions, QKV GEMM (64x64 tiles/wave), flash
// attention (64 queries/wave, K/V operand reuse across 4 q-tiles), out GEMM.
// ---------------------------------------------------------------------------

typedef __attribute__((ext_vector_type(16))) __bf16 v16bf;
typedef __attribute__((ext_vector_type(8)))  __bf16 v8bf;
typedef __attribute__((ext_vector_type(8)))  float  v8f;

#define LOG2E 1.4426950408889634f

__device__ __forceinline__ v8f wmma_bf(v16bf a, v16bf b, v8f c) {
    return __builtin_amdgcn_wmma_f32_16x16x32_bf16(false, a, false, b,
                                                   (short)0, c, false, false);
}

// Load a 16x32 A-operand row chunk. Lane layout (contiguous, matches B):
//   elements 0..7  : K = 2*kb + 0..7    (kb = (lane>>4)*8)
//   elements 8..15 : K = 2*kb + 8 + 0..7
// `p` points at (row, k0) of a row-major bf16 matrix.
__device__ __forceinline__ v16bf load_a16(const __bf16* p, int kbA) {
    v8bf lo = *(const v8bf*)(p + 2 * kbA);
    v8bf hi = *(const v8bf*)(p + 2 * kbA + 8);
    v16bf r;
#pragma unroll
    for (int i = 0; i < 8; ++i) { r[i] = lo[i]; r[i + 8] = hi[i]; }
    return r;
}

// ---------------------------------------------------------------------------
// Conversion kernels
// ---------------------------------------------------------------------------
__global__ void cvt_f32_bf16(const float* __restrict__ in, __bf16* __restrict__ out, int n) {
    int i = blockIdx.x * blockDim.x + threadIdx.x;
    if (i < n) out[i] = (__bf16)in[i];
}

// in[rows][cols] (f32, row-major) -> out[cols][rows] (bf16)
__global__ void transpose_f32_bf16(const float* __restrict__ in, __bf16* __restrict__ out,
                                   int rows, int cols) {
    int i = blockIdx.x * blockDim.x + threadIdx.x;
    if (i < rows * cols) {
        int r = i / cols, c = i % cols;
        out[(size_t)c * rows + r] = (__bf16)in[i];
    }
}

// ---------------------------------------------------------------------------
// QKV projection: D[4096,3072] = Xbf[4096,1024] * Wqkv  (Bt = Wqkv^T, row-major)
// Each wave computes a 64x64 tile; `part` (Q/K/V) is wave-uniform since the
// 1024-wide Q/K/V column bands are multiples of the 64-wide tile.
// ---------------------------------------------------------------------------
__global__ __launch_bounds__(256)
void gemm_qkv_kernel(const __bf16* __restrict__ A, const __bf16* __restrict__ Bt,
                     __bf16* __restrict__ Qo, __bf16* __restrict__ Ko,
                     __bf16* __restrict__ Vto) {
    const int lane = threadIdx.x & 31;
    const int wid  = blockIdx.x * (blockDim.x >> 5) + (threadIdx.x >> 5);
    const int tilesN = 3072 / 64;                // 48
    const int tm = wid / tilesN, tn = wid % tilesN;
    const int lm = lane & 15, hi = lane >> 4;
    const int kbA = hi * 8, kbB = hi * 16;

    v8f acc[4][4] = {};
    const __bf16* Arow[4];
    const __bf16* Brow[4];
#pragma unroll
    for (int i = 0; i < 4; ++i) Arow[i] = A  + (size_t)(tm * 64 + 16 * i + lm) * 1024;
#pragma unroll
    for (int j = 0; j < 4; ++j) Brow[j] = Bt + (size_t)(tn * 64 + 16 * j + lm) * 1024;

    for (int kk = 0; kk < 1024; kk += 32) {
        v16bf a[4], b[4];
#pragma unroll
        for (int i = 0; i < 4; ++i) a[i] = load_a16(Arow[i] + kk, kbA);
#pragma unroll
        for (int j = 0; j < 4; ++j) b[j] = *(const v16bf*)(Brow[j] + kk + kbB);
#pragma unroll
        for (int i = 0; i < 4; ++i)
#pragma unroll
            for (int j = 0; j < 4; ++j) acc[i][j] = wmma_bf(a[i], b[j], acc[i][j]);
    }

    // Wave-uniform destination select: tn in [0,16)->Q, [16,32)->K, [32,48)->V
    const int part = tn >> 4;
    const int cn   = tn & 15;                    // head index
    if (part == 0) {
#pragma unroll
        for (int i = 0; i < 4; ++i)
#pragma unroll
            for (int j = 0; j < 4; ++j)
#pragma unroll
                for (int r = 0; r < 8; ++r) {
                    int row = tm * 64 + 16 * i + r + 8 * hi;
                    int bb = row >> 11, t = row & 2047, d = 16 * j + lm;
                    Qo[(((size_t)(bb * 16 + cn)) * 2048 + t) * 64 + d] = (__bf16)acc[i][j][r];
                }
    } else if (part == 1) {
#pragma unroll
        for (int i = 0; i < 4; ++i)
#pragma unroll
            for (int j = 0; j < 4; ++j)
#pragma unroll
                for (int r = 0; r < 8; ++r) {
                    int row = tm * 64 + 16 * i + r + 8 * hi;
                    int bb = row >> 11, t = row & 2047, d = 16 * j + lm;
                    Ko[(((size_t)(bb * 16 + cn)) * 2048 + t) * 64 + d] = (__bf16)acc[i][j][r];
                }
    } else {
#pragma unroll
        for (int i = 0; i < 4; ++i)
#pragma unroll
            for (int j = 0; j < 4; ++j)
#pragma unroll
                for (int r = 0; r < 8; ++r) {
                    int row = tm * 64 + 16 * i + r + 8 * hi;
                    int bb = row >> 11, t = row & 2047, d = 16 * j + lm;
                    Vto[(((size_t)(bb * 16 + cn)) * 64 + d) * 2048 + t] = (__bf16)acc[i][j][r];
                }
    }
}

// ---------------------------------------------------------------------------
// Flash attention: one wave per (b,h,64-query block). Four 16-row q-tiles per
// wave share every K/V operand load (4x arithmetic intensity vs 1 tile/wave).
// Online softmax, causal mask computed analytically.
// ---------------------------------------------------------------------------
__global__ __launch_bounds__(32)
void attn_kernel(const __bf16* __restrict__ Q, const __bf16* __restrict__ K,
                 const __bf16* __restrict__ Vt, __bf16* __restrict__ O) {
    __shared__ __bf16 pLds[4 * 16 * 32];        // one 16x32 P tile per q-tile

    const int lane = threadIdx.x & 31;
    const int lm = lane & 15, hi = lane >> 4;
    const int kbA = hi * 8, kbB = hi * 16;

    const int bh = blockIdx.x >> 5;             // T/64 = 32 q-blocks per (b,h)
    const int qb = blockIdx.x & 31;
    const int q0 = qb * 64;
    const int b  = bh >> 4, h = bh & 15;

    const __bf16* Qbase = Q  + ((size_t)bh * 2048 + q0) * 64;
    const __bf16* Kbase = K  + (size_t)bh * 2048 * 64;
    const __bf16* Vbase = Vt + (size_t)bh * 64 * 2048;

    // Q tiles as A-operands: qa[qi][0] d=0..31, qa[qi][1] d=32..63
    v16bf qa[4][2];
#pragma unroll
    for (int qi = 0; qi < 4; ++qi) {
        const __bf16* Qp = Qbase + (size_t)(16 * qi + lm) * 64;
        qa[qi][0] = load_a16(Qp, kbA);
        qa[qi][1] = load_a16(Qp + 32, kbA);
    }

    v8f acc[4][4] = {};              // [q-tile][d-subtile], cols d = 16n + lm
    float m[4][8], l[4][8];
#pragma unroll
    for (int qi = 0; qi < 4; ++qi)
#pragma unroll
        for (int r = 0; r < 8; ++r) { m[qi][r] = -1e30f; l[qi][r] = 0.f; }
    const float scale = 0.125f;      // 1/sqrt(64)

    const int qend = q0 + 64;
    for (int j0 = 0; j0 < qend; j0 += 32) {
        __syncthreads();             // protect LDS reuse from previous iter

        // ---- shared K operands for keys [j0, j0+32) ----
        const __bf16* Kp0 = Kbase + (size_t)(j0 + lm) * 64;
        const __bf16* Kp1 = Kbase + (size_t)(j0 + 16 + lm) * 64;
        v16bf b00 = *(const v16bf*)(Kp0 + kbB);
        v16bf b01 = *(const v16bf*)(Kp0 + 32 + kbB);
        v16bf b10 = *(const v16bf*)(Kp1 + kbB);
        v16bf b11 = *(const v16bf*)(Kp1 + 32 + kbB);

        // ---- S = Q*K^T for all four q-tiles ----
        v8f s0[4], s1[4];
#pragma unroll
        for (int qi = 0; qi < 4; ++qi) {
            v8f z = {};
            s0[qi] = wmma_bf(qa[qi][0], b00, z); s0[qi] = wmma_bf(qa[qi][1], b01, s0[qi]);
            s1[qi] = wmma_bf(qa[qi][0], b10, z); s1[qi] = wmma_bf(qa[qi][1], b11, s1[qi]);
        }

        // ---- mask + online softmax + acc rescale + P->LDS, per q-tile ----
#pragma unroll
        for (int qi = 0; qi < 4; ++qi) {
            float fac[8];
#pragma unroll
            for (int r = 0; r < 8; ++r) {
                int row  = q0 + 16 * qi + r + 8 * hi;
                int key0 = j0 + lm, key1 = j0 + 16 + lm;
                float v0 = (key0 <= row) ? s0[qi][r] * scale : -1e30f;
                float v1 = (key1 <= row) ? s1[qi][r] * scale : -1e30f;
                float mx = fmaxf(v0, v1);
#pragma unroll
                for (int off = 8; off; off >>= 1) mx = fmaxf(mx, __shfl_xor(mx, off));
                float mnew = fmaxf(m[qi][r], mx);
                fac[r] = exp2f((m[qi][r] - mnew) * LOG2E);
                float p0v = exp2f((v0 - mnew) * LOG2E);
                float p1v = exp2f((v1 - mnew) * LOG2E);
                s0[qi][r] = p0v; s1[qi][r] = p1v;   // reuse S regs for P
                float rs = p0v + p1v;
#pragma unroll
                for (int off = 8; off; off >>= 1) rs += __shfl_xor(rs, off);
                l[qi][r] = l[qi][r] * fac[r] + rs;
                m[qi][r] = mnew;
            }
#pragma unroll
            for (int n = 0; n < 4; ++n)
#pragma unroll
                for (int r = 0; r < 8; ++r) acc[qi][n][r] *= fac[r];
#pragma unroll
            for (int r = 0; r < 8; ++r) {
                int row = r + 8 * hi;
                pLds[qi * 512 + row * 32 + lm]      = (__bf16)s0[qi][r];
                pLds[qi * 512 + row * 32 + 16 + lm] = (__bf16)s1[qi][r];
            }
        }
        __syncthreads();

        // ---- P back as A-operands; shared V operands; O += P*V ----
        v16bf pa[4];
#pragma unroll
        for (int qi = 0; qi < 4; ++qi) pa[qi] = load_a16(&pLds[qi * 512 + lm * 32], kbA);

        v16bf vv[4];
#pragma unroll
        for (int n = 0; n < 4; ++n)
            vv[n] = *(const v16bf*)(Vbase + (size_t)(16 * n + lm) * 2048 + j0 + kbB);

#pragma unroll
        for (int qi = 0; qi < 4; ++qi)
#pragma unroll
            for (int n = 0; n < 4; ++n) acc[qi][n] = wmma_bf(pa[qi], vv[n], acc[qi][n]);
    }

    // ---- normalize and store to O[b][t][h*64+d] ----
#pragma unroll
    for (int qi = 0; qi < 4; ++qi) {
        float invl[8];
#pragma unroll
        for (int r = 0; r < 8; ++r) invl[r] = 1.f / l[qi][r];
#pragma unroll
        for (int n = 0; n < 4; ++n)
#pragma unroll
            for (int r = 0; r < 8; ++r) {
                int q = q0 + 16 * qi + r + 8 * hi;
                int d = 16 * n + lm;
                O[((size_t)b * 2048 + q) * 1024 + h * 64 + d] =
                    (__bf16)(acc[qi][n][r] * invl[r]);
            }
    }
}

// ---------------------------------------------------------------------------
// Output projection: out[4096,1024] (f32) = Obf[4096,1024] * Wout (Bt = Wout^T)
// ---------------------------------------------------------------------------
__global__ __launch_bounds__(256)
void gemm_out_kernel(const __bf16* __restrict__ A, const __bf16* __restrict__ Bt,
                     float* __restrict__ out) {
    const int lane = threadIdx.x & 31;
    const int wid  = blockIdx.x * (blockDim.x >> 5) + (threadIdx.x >> 5);
    const int tilesN = 1024 / 64;                // 16
    const int tm = wid / tilesN, tn = wid % tilesN;
    const int lm = lane & 15, hi = lane >> 4;
    const int kbA = hi * 8, kbB = hi * 16;

    v8f acc[4][4] = {};
    const __bf16* Arow[4];
    const __bf16* Brow[4];
#pragma unroll
    for (int i = 0; i < 4; ++i) Arow[i] = A  + (size_t)(tm * 64 + 16 * i + lm) * 1024;
#pragma unroll
    for (int j = 0; j < 4; ++j) Brow[j] = Bt + (size_t)(tn * 64 + 16 * j + lm) * 1024;

    for (int kk = 0; kk < 1024; kk += 32) {
        v16bf a[4], b[4];
#pragma unroll
        for (int i = 0; i < 4; ++i) a[i] = load_a16(Arow[i] + kk, kbA);
#pragma unroll
        for (int j = 0; j < 4; ++j) b[j] = *(const v16bf*)(Brow[j] + kk + kbB);
#pragma unroll
        for (int i = 0; i < 4; ++i)
#pragma unroll
            for (int j = 0; j < 4; ++j) acc[i][j] = wmma_bf(a[i], b[j], acc[i][j]);
    }

#pragma unroll
    for (int i = 0; i < 4; ++i)
#pragma unroll
        for (int j = 0; j < 4; ++j)
#pragma unroll
            for (int r = 0; r < 8; ++r) {
                int row = tm * 64 + 16 * i + r + 8 * hi;
                int col = tn * 64 + 16 * j + lm;
                out[(size_t)row * 1024 + col] = acc[i][j][r];
            }
}

// ---------------------------------------------------------------------------
// Launch
// ---------------------------------------------------------------------------
extern "C" void kernel_launch(void* const* d_in, const int* in_sizes, int n_in,
                              void* d_out, int out_size, void* d_ws, size_t ws_size,
                              hipStream_t stream) {
    const float* x    = (const float*)d_in[0];   // [2,2048,1024]
    // d_in[1] = mask (causal tril) — computed analytically in the kernel
    const float* Wqkv = (const float*)d_in[2];   // [1024,3072]
    const float* Wout = (const float*)d_in[3];   // [1024,1024]
    float* out = (float*)d_out;                  // [2,2048,1024]

    char* ws = (char*)d_ws;
    __bf16* Xbf   = (__bf16*)(ws + 0);           // 4096*1024 bf16  (8 MB)
    __bf16* WqkvT = (__bf16*)(ws + (size_t)8  * 1024 * 1024);  // [3072,1024] (6 MB)
    __bf16* WoutT = (__bf16*)(ws + (size_t)14 * 1024 * 1024);  // [1024,1024] (2 MB)
    __bf16* Qbf   = (__bf16*)(ws + (size_t)16 * 1024 * 1024);  // [B,H,T,64]  (8 MB)
    __bf16* Kbf   = (__bf16*)(ws + (size_t)24 * 1024 * 1024);  // [B,H,T,64]  (8 MB)
    __bf16* Vt    = (__bf16*)(ws + (size_t)32 * 1024 * 1024);  // [B,H,64,T]  (8 MB)
    __bf16* Obf   = (__bf16*)(ws + (size_t)40 * 1024 * 1024);  // [B,T,C]     (8 MB)

    const int nX = 4096 * 1024;
    cvt_f32_bf16<<<nX / 256, 256, 0, stream>>>(x, Xbf, nX);

    const int nWq = 1024 * 3072;
    transpose_f32_bf16<<<(nWq + 255) / 256, 256, 0, stream>>>(Wqkv, WqkvT, 1024, 3072);
    const int nWo = 1024 * 1024;
    transpose_f32_bf16<<<(nWo + 255) / 256, 256, 0, stream>>>(Wout, WoutT, 1024, 1024);

    // QKV GEMM: 64 x 48 = 3072 wave-tiles, 8 waves per 256-thread block
    gemm_qkv_kernel<<<3072 / 8, 256, 0, stream>>>(Xbf, WqkvT, Qbf, Kbf, Vt);

    // Flash attention: B*H*(T/64) = 1024 single-wave blocks, 64 queries each
    attn_kernel<<<1024, 32, 0, stream>>>(Qbf, Kbf, Vt, Obf);

    // Output projection: 64 x 16 = 1024 wave-tiles
    gemm_out_kernel<<<1024 / 8, 256, 0, stream>>>(Obf, WoutT, out);
}
